// ConstraintAwareLoss_16131897164169
// MI455X (gfx1250) — compile-verified, matrix operand order checked by reading, not run on hardware
//
#include <hip/hip_runtime.h>
#include <hip/hip_bf16.h>
#include <hip/hip_fp16.h>

typedef __attribute__((ext_vector_type(16))) _Float16 v16h;
typedef __attribute__((ext_vector_type(8)))  float    v8f;

#define N_TASKS 4096
#define N_PROC  64
#define N_EDGES 65536
#define TILE    256

// ws layout (floats): [0]=procloss sum, [1]=start sum, [2]=end sum, [3]=dur sum,
// [4]=precedence sum, [5]=overlap sum, [6]=count (as raw u32 bits), [7]=pad,
// ws+16 .. : int proc[4096]

__global__ void zero_kernel(float* ws) {
    if (threadIdx.x < 8) ws[threadIdx.x] = 0.0f;  // also zeroes the u32 count at [6]
}

// One wave handles 16 rows of the 4096x64 matrix.
// WMMA computes sum(exp(x - rowmax)) over each row: A(16x32 f16) x ones(32x16), f32 acc.
__global__ __launch_bounds__(256) void softmax_argmax_kernel(
    const float* __restrict__ processor, const int* __restrict__ y_proc,
    int* __restrict__ proc_out, float* __restrict__ acc_procloss)
{
    __shared__ float s_sumexp[8][16];
    const int tid  = threadIdx.x;
    const int wv   = tid >> 5;
    const int lane = tid & 31;
    const int m    = lane & 15;      // row within 16-row tile (A layout: M = lane%16)
    const int g    = lane >> 4;      // lane group
    const int row  = blockIdx.x * 128 + wv * 16 + m;
    const float* rp = processor + row * N_PROC;

    // Lane loads cols [32c + 16g, 32c + 16g + 16) for chunks c=0,1 (vectorized).
    float xs[2][16];
    #pragma unroll
    for (int c = 0; c < 2; ++c) {
        const float4* p4 = (const float4*)(rp + 32 * c + 16 * g);
        #pragma unroll
        for (int q = 0; q < 4; ++q) {
            float4 v = p4[q];
            xs[c][4*q+0] = v.x; xs[c][4*q+1] = v.y;
            xs[c][4*q+2] = v.z; xs[c][4*q+3] = v.w;
        }
    }

    // Per-lane max + argmax (cols increase within a lane -> strict > keeps first occurrence).
    float bv = xs[0][0]; int bi = 16 * g;
    #pragma unroll
    for (int c = 0; c < 2; ++c)
        #pragma unroll
        for (int t = 0; t < 16; ++t) {
            float v = xs[c][t];
            int col = 32 * c + 16 * g + t;
            if (v > bv) { bv = v; bi = col; }
        }
    // Combine lanes m and m+16 (symmetric: both end with row result).
    float ov = __shfl_xor(bv, 16);
    int   oi = __shfl_xor(bi, 16);
    if (ov > bv || (ov == bv && oi < bi)) { bv = ov; bi = oi; }
    const float rowmax = bv;

    // Build A = exp(x - rowmax) in f16, B = ones; accumulate row sums via WMMA.
    v16h a0, a1, bones;
    #pragma unroll
    for (int t = 0; t < 16; ++t) {
        a0[t] = (_Float16)expf(xs[0][t] - rowmax);
        a1[t] = (_Float16)expf(xs[1][t] - rowmax);
        bones[t] = (_Float16)1.0f;
    }
    v8f acc = {};
    acc = __builtin_amdgcn_wmma_f32_16x16x32_f16(false, a0, false, bones,
                                                 (short)0, acc, false, false);
    acc = __builtin_amdgcn_wmma_f32_16x16x32_f16(false, a1, false, bones,
                                                 (short)0, acc, false, false);

    // D[m][n] = sumexp[m] for all n. Lane 0 holds rows 0-7 (vgpr r), lane 16 rows 8-15.
    if (lane == 0 || lane == 16) {
        #pragma unroll
        for (int r = 0; r < 8; ++r) s_sumexp[wv][8 * g + r] = acc[r];
    }
    __syncthreads();

    float contrib = 0.0f;
    if (lane < 16) {
        float se = s_sumexp[wv][m];
        int y = y_proc[row];
        float xv = rp[y];
        contrib = rowmax + logf(se) - xv;   // -(logp at y)
        proc_out[row] = bi;                  // argmax for the overlap mask
    }
    #pragma unroll
    for (int off = 16; off > 0; off >>= 1) contrib += __shfl_xor(contrib, off);
    if (lane == 0) atomicAdd(acc_procloss, contrib);
}

// 4096x4096 masked pairwise overlap; grid (16,16) of 256x256 tiles, upper-tri blocks only.
__global__ __launch_bounds__(256) void overlap_kernel(
    const float* __restrict__ s, const float* __restrict__ e,
    const int* __restrict__ proc, float* __restrict__ accS, unsigned* __restrict__ accC)
{
    const int it = blockIdx.y, jt = blockIdx.x;
    if (jt < it) return;
    __shared__ float sj[TILE];
    __shared__ float ej[TILE];
    __shared__ int   pj[TILE];
    const int tid = threadIdx.x;
    const int gj0 = jt * TILE;
    sj[tid] = s[gj0 + tid];
    ej[tid] = e[gj0 + tid];
    pj[tid] = proc[gj0 + tid];
    __syncthreads();

    const int gi = it * TILE + tid;
    const float si_ = s[gi], ei_ = e[gi];
    const int pi_ = proc[gi];
    float sum = 0.0f;
    unsigned cnt = 0;
    #pragma unroll 4
    for (int jj = 0; jj < TILE; ++jj) {
        int gjj = gj0 + jj;
        if (gjj > gi && pj[jj] == pi_) {
            float w = fminf(ei_, ej[jj]) - fmaxf(si_, sj[jj]);
            sum += fmaxf(w, 0.0f);
            cnt++;                       // mask count includes zero-overlap pairs
        }
    }
    #pragma unroll
    for (int off = 16; off > 0; off >>= 1) {
        sum += __shfl_xor(sum, off);
        cnt += (unsigned)__shfl_xor((int)cnt, off);
    }
    if ((tid & 31) == 0) {
        atomicAdd(accS, sum);
        atomicAdd(accC, cnt);
    }
}

__global__ void edge_kernel(const int* __restrict__ edges,
                            const float* __restrict__ s, const float* __restrict__ e,
                            float* __restrict__ accP)
{
    int k = blockIdx.x * blockDim.x + threadIdx.x;
    float v = 0.0f;
    if (k < N_EDGES) {
        int snd = edges[k];
        int rcv = edges[N_EDGES + k];
        v = fmaxf(e[snd] - s[rcv], 0.0f);
    }
    #pragma unroll
    for (int off = 16; off > 0; off >>= 1) v += __shfl_xor(v, off);
    if ((threadIdx.x & 31) == 0) atomicAdd(accP, v);
}

__global__ void l1_kernel(const float* __restrict__ s, const float* __restrict__ e,
                          const float* __restrict__ dur, const float* __restrict__ ys,
                          const float* __restrict__ ye, float* __restrict__ ws)
{
    int i = blockIdx.x * blockDim.x + threadIdx.x;
    float a = 0.0f, b = 0.0f, c = 0.0f;
    if (i < N_TASKS) {
        float si = s[i], ei = e[i];
        a = fabsf(si - ys[i]);
        b = fabsf(ei - ye[i]);
        c = fabsf((ei - si) - dur[i]);
    }
    #pragma unroll
    for (int off = 16; off > 0; off >>= 1) {
        a += __shfl_xor(a, off);
        b += __shfl_xor(b, off);
        c += __shfl_xor(c, off);
    }
    if ((threadIdx.x & 31) == 0) {
        atomicAdd(ws + 1, a);
        atomicAdd(ws + 2, b);
        atomicAdd(ws + 3, c);
    }
}

__global__ void finalize_kernel(const float* __restrict__ ws,
                                const float* __restrict__ makespan,
                                const float* __restrict__ y_makespan,
                                float* __restrict__ out)
{
    if (threadIdx.x == 0 && blockIdx.x == 0) {
        float proc_loss  = ws[0] * (1.0f / N_TASKS);
        float start_loss = ws[1] * (1.0f / N_TASKS);
        float end_loss   = ws[2] * (1.0f / N_TASKS);
        float dur_loss   = ws[3] * (1.0f / N_TASKS);
        float prec       = ws[4] * (1.0f / N_EDGES);
        unsigned cnt = ((const unsigned*)ws)[6];
        float ovl = ws[5] / (float)(cnt > 1u ? cnt : 1u);
        float make = fabsf(makespan[0] - y_makespan[0]);
        out[0] = proc_loss + start_loss + end_loss + make
               + 10.0f * prec + 10.0f * ovl + 5.0f * dur_loss;
    }
}

extern "C" void kernel_launch(void* const* d_in, const int* in_sizes, int n_in,
                              void* d_out, int out_size, void* d_ws, size_t ws_size,
                              hipStream_t stream) {
    const float* processor  = (const float*)d_in[0];
    const float* start_t    = (const float*)d_in[1];
    const float* end_t      = (const float*)d_in[2];
    const float* duration   = (const float*)d_in[3];
    const float* makespan   = (const float*)d_in[4];
    const float* y_start    = (const float*)d_in[5];
    const float* y_end      = (const float*)d_in[6];
    const float* y_makespan = (const float*)d_in[7];
    const int*   edges      = (const int*)d_in[8];
    const int*   y_proc     = (const int*)d_in[9];

    float* ws   = (float*)d_ws;
    int*   proc = (int*)(ws + 16);
    float* out  = (float*)d_out;

    zero_kernel<<<1, 32, 0, stream>>>(ws);
    softmax_argmax_kernel<<<32, 256, 0, stream>>>(processor, y_proc, proc, ws);
    overlap_kernel<<<dim3(16, 16), 256, 0, stream>>>(start_t, end_t, proc,
                                                     ws + 5, (unsigned*)ws + 6);
    edge_kernel<<<N_EDGES / 256, 256, 0, stream>>>(edges, end_t, start_t, ws + 4);
    l1_kernel<<<N_TASKS / 256, 256, 0, stream>>>(start_t, end_t, duration,
                                                 y_start, y_end, ws);
    finalize_kernel<<<1, 1, 0, stream>>>(ws, makespan, y_makespan, out);
}
